// O3AttentionLayer_16836271800692
// MI455X (gfx1250) — matrix operand order repaired and checked by reading, not compiled, not run on hardware
//
#include <hip/hip_runtime.h>
#include <hip/hip_bf16.h>
#include <math.h>

// ---------------------------------------------------------------------------
// O3 equivariant attention, MI455X (gfx1250, wave32, WMMA f16 16x16x32).
// Fused per-edge basis-MLP + tensor product on v_wmma_f32_16x16x32_f16 with
// f16 hi/lo splitting (3 WMMAs) for near-fp32 accuracy. Per-edge 576-wide
// weight tensors are never materialized to global memory.
// Workspace need: ~14.5 MB.
// ---------------------------------------------------------------------------

#define N_NODES 10000
#define N_EDGES 160000
#define NTILES  (N_EDGES / 16)

typedef _Float16 half16_t __attribute__((ext_vector_type(16)));
typedef _Float16 half8_t  __attribute__((ext_vector_type(8)));
typedef float    float8_t __attribute__((ext_vector_type(8)));

union V16 { half16_t v; half8_t h2[2]; _Float16 e[16]; };
union C8  { float8_t  v; float e[8]; };

__device__ __forceinline__ float8_t wmma_f16(half16_t a, half16_t b, float8_t c) {
  // D = A(16x32 f16) * B(32x16 f16) + C(16x16 f32)
  return __builtin_amdgcn_wmma_f32_16x16x32_f16(false, a, false, b, (short)0, c,
                                                false, false);
}

// Intra-wave LDS phase fence: per-wave DS ops are processed in order, so a
// compile-time barrier + dscnt drain is sufficient (no block barrier needed).
__device__ __forceinline__ void lds_fence() {
  asm volatile("s_wait_dscnt 0" ::: "memory");
}

struct __align__(16) WaveScr {
  float h[16][32];       // silu(rb@W1), [edge][hidden]
  float a0[32][16];      // x0[src],           [u][edge]
  float a110[16][16];    // (x1 . vec)/r,      [u][edge]
  float x1c[16][3][16];  // x1[src],           [u][m][edge]
  float acc[2][16][24];  // per net: [edge][ k0(8) | t01(4) | t10(4*3) ]
  float simp[16];        // 1o similarity partial per edge
};

struct __align__(16) BlockShared {
  _Float16 w1hi[2][32 * 32];   // [net][col*32 + k]  (B layout: col-major rows)
  _Float16 w1lo[2][32 * 32];
  _Float16 w2hi[2][576 * 32];  // [net][col*32 + k], includes 1/sqrt(32*48)
  _Float16 w2lo[2][576 * 32];
  float wsim0[64];
  float wsim1[16];
  WaveScr ws[8];
};
static_assert(sizeof(BlockShared) <= 320 * 1024, "exceeds WGP LDS");

__launch_bounds__(256, 1)
__global__ void edge_kernel(const float* __restrict__ x,
                            const float* __restrict__ pos,
                            const float* __restrict__ w_kb1,
                            const float* __restrict__ w_kb2,
                            const float* __restrict__ w_vb1,
                            const float* __restrict__ w_vb2,
                            const float* __restrict__ w_sim0,
                            const float* __restrict__ w_sim1,
                            const int*   __restrict__ ei,
                            const float* __restrict__ qbuf,
                            float* __restrict__ logits,
                            float* __restrict__ vals) {
  __shared__ BlockShared sm;
  const int tid = threadIdx.x;

  // ---- one-time per block: split weights to f16 hi/lo in B-friendly layout
  for (int i = tid; i < 32 * 32; i += 256) {
    int k = i >> 5, n = i & 31;
    float wk = w_kb1[i], wv = w_vb1[i];
    _Float16 hk = (_Float16)wk, hv = (_Float16)wv;
    sm.w1hi[0][n * 32 + k] = hk;
    sm.w1lo[0][n * 32 + k] = (_Float16)(wk - (float)hk);
    sm.w1hi[1][n * 32 + k] = hv;
    sm.w1lo[1][n * 32 + k] = (_Float16)(wv - (float)hv);
  }
  const float SCL = 0.025515518153991442f;  // 1/sqrt(32*48): hidden norm * TP fan-in
  for (int i = tid; i < 32 * 576; i += 256) {
    int k = i / 576, n = i % 576;
    float wk = w_kb2[i] * SCL, wv = w_vb2[i] * SCL;
    _Float16 hk = (_Float16)wk, hv = (_Float16)wv;
    sm.w2hi[0][n * 32 + k] = hk;
    sm.w2lo[0][n * 32 + k] = (_Float16)(wk - (float)hk);
    sm.w2hi[1][n * 32 + k] = hv;
    sm.w2lo[1][n * 32 + k] = (_Float16)(wv - (float)hv);
  }
  if (tid < 64) sm.wsim0[tid] = w_sim0[tid];
  if (tid < 16) sm.wsim1[tid] = w_sim1[tid];
  __syncthreads();

  const int lane = tid & 31;
  const int wv_  = tid >> 5;
  const int half = lane >> 4;   // A/B K-group select (K bit4)
  const int eL   = lane & 15;   // edge-in-tile / column-in-tile
  WaveScr& W = sm.ws[wv_];

  const float RMAXI = 1.0f / 2.5f;
  const float SQ3   = 1.7320508075688772f;
  const float CS    = 0.11180339887498948f;  // 1/sqrt(80)

  for (int T = blockIdx.x * 8 + wv_; T < NTILES; T += gridDim.x * 8) {
    const int e = T * 16 + eL;
    const int s = ei[e];
    const int d = ei[N_EDGES + e];
    const float vx = pos[s * 3 + 0] - pos[d * 3 + 0];
    const float vy = pos[s * 3 + 1] - pos[d * 3 + 1];
    const float vz = pos[s * 3 + 2] - pos[d * 3 + 2];
    const float r  = sqrtf(vx * vx + vy * vy + vz * vz + 1e-24f);
    const float rinv = 1.0f / r;
    float tcut = 10.0f * (1.0f - r * RMAXI);
    const float cut = (tcut > 0.0f) ? __expf(-1.0f / fmaxf(tcut, 1e-12f)) : 0.0f;

    // ---- gather node features for this tile (2 lanes per edge)
    float qreg[12] = {0};
    if (half == 0) {
      const float4* xr = (const float4*)(x + (long)s * 80);
#pragma unroll
      for (int j = 0; j < 8; ++j) {
        float4 v4 = xr[j];
        W.a0[j * 4 + 0][eL] = v4.x; W.a0[j * 4 + 1][eL] = v4.y;
        W.a0[j * 4 + 2][eL] = v4.z; W.a0[j * 4 + 3][eL] = v4.w;
      }
      const float4* qr = (const float4*)(qbuf + (long)s * 20);
      float4 qa = qr[0], qb = qr[1];
      qreg[0] = qa.x; qreg[1] = qa.y; qreg[2] = qa.z; qreg[3] = qa.w;
      qreg[4] = qb.x; qreg[5] = qb.y; qreg[6] = qb.z; qreg[7] = qb.w;
    } else {
      const float4* xr = (const float4*)(x + (long)s * 80 + 32);
      float X1[48];
#pragma unroll
      for (int j = 0; j < 12; ++j) {
        float4 v4 = xr[j];
        X1[j * 4 + 0] = v4.x; X1[j * 4 + 1] = v4.y;
        X1[j * 4 + 2] = v4.z; X1[j * 4 + 3] = v4.w;
      }
#pragma unroll
      for (int u = 0; u < 16; ++u) {
        float a = X1[u * 3 + 0], b = X1[u * 3 + 1], c = X1[u * 3 + 2];
        W.x1c[u][0][eL] = a; W.x1c[u][1][eL] = b; W.x1c[u][2][eL] = c;
        W.a110[u][eL] = (a * vx + b * vy + c * vz) * rinv;  // x1.Y / sqrt3
      }
      const float4* qr = (const float4*)(qbuf + (long)s * 20 + 8);
      float4 qa = qr[0], qb = qr[1], qc = qr[2];
      qreg[0] = qa.x; qreg[1]  = qa.y; qreg[2]  = qa.z; qreg[3]  = qa.w;
      qreg[4] = qb.x; qreg[5]  = qb.y; qreg[6]  = qb.z; qreg[7]  = qb.w;
      qreg[8] = qc.x; qreg[9]  = qc.y; qreg[10] = qc.z; qreg[11] = qc.w;
    }
    {  // zero TP accumulators
      float* az = (float*)W.acc;
      for (int i = lane; i < 2 * 16 * 24; i += 32) az[i] = 0.0f;
    }
    lds_fence();

    // ---- rb in WMMA A layout (hi/lo split), Bessel via sin recurrence
    V16 Rhi, Rlo;
    {
      float th = 3.14159265358979f * r * RMAXI;
      float s1, c1;
      __sincosf(th, &s1, &c1);
      const float pref = 0.8944271909999159f * rinv;  // sqrt(2/2.5)/r
      float sk = s1, skm1 = 0.0f;
      const float twoc = 2.0f * c1;
#pragma unroll
      for (int k = 0; k < 32; ++k) {
        if ((k >> 4) == half) {  // lane half owns K with matching bit4
          const int p = k & 15;
          float val = pref * sk;
          _Float16 hi = (_Float16)val;
          Rhi.e[p] = hi;
          Rlo.e[p] = (_Float16)(val - (float)hi);
        }
        float nxt = twoc * sk - skm1;
        skm1 = sk; sk = nxt;
      }
    }

#pragma unroll 1
    for (int net = 0; net < 2; ++net) {  // 0 = key, 1 = value
      // ---- GEMM1: h = silu(rb @ W1), 16 edges x 32 hidden
      const half8_t* B1h = (const half8_t*)sm.w1hi[net];
      const half8_t* B1l = (const half8_t*)sm.w1lo[net];
#pragma unroll
      for (int tb = 0; tb < 2; ++tb) {
        const int bi = (tb * 16 + eL) * 4 + half * 2;
        V16 Bh, Bl;
        Bh.h2[0] = B1h[bi]; Bh.h2[1] = B1h[bi + 1];
        Bl.h2[0] = B1l[bi]; Bl.h2[1] = B1l[bi + 1];
        C8 C;
#pragma unroll
        for (int i = 0; i < 8; ++i) C.e[i] = 0.0f;
        C.v = wmma_f16(Rlo.v, Bh.v, C.v);
        C.v = wmma_f16(Rhi.v, Bl.v, C.v);
        C.v = wmma_f16(Rhi.v, Bh.v, C.v);
#pragma unroll
        for (int rr = 0; rr < 8; ++rr) {
          float hv = C.e[rr];
          // silu via hardware v_rcp_f32 (1 ulp) -- avoids IEEE divide expansion
          hv = hv * __builtin_amdgcn_rcpf(1.0f + __expf(-hv));
          W.h[rr + 8 * half][tb * 16 + eL] = hv;
        }
      }
      lds_fence();

      // ---- GEMM2 A-operand (hi/lo) from h
      V16 Hhi, Hlo;
      {
        const float4* hr = (const float4*)(&W.h[eL][0]);
        const int i0 = half * 4;
        float4 p0 = hr[i0], p1 = hr[i0 + 1], p2 = hr[i0 + 2], p3 = hr[i0 + 3];
        float tmp[16] = {p0.x, p0.y, p0.z, p0.w, p1.x, p1.y, p1.z, p1.w,
                         p2.x, p2.y, p2.z, p2.w, p3.x, p3.y, p3.z, p3.w};
#pragma unroll
        for (int i = 0; i < 16; ++i) {
          _Float16 hi = (_Float16)tmp[i];
          Hhi.e[i] = hi;
          Hlo.e[i] = (_Float16)(tmp[i] - (float)hi);
        }
      }

      const half8_t* B2h = (const half8_t*)sm.w2hi[net];
      const half8_t* B2l = (const half8_t*)sm.w2lo[net];
      float* accE = (float*)&W.acc[net][0][0];
      const int w8 = eL & 7;   // output channel for 0e paths (const per lane)
      const int w4 = eL & 3;   // output channel for 1o paths

      // ---- region 00 (tiles 0..15) + 110 (16..23): k0 accumulation
      float p00[8];
#pragma unroll
      for (int i = 0; i < 8; ++i) p00[i] = 0.0f;
#pragma unroll 2
      for (int tt = 0; tt < 16; ++tt) {
        const int bi = (tt * 16 + eL) * 4 + half * 2;
        V16 Bh, Bl;
        Bh.h2[0] = B2h[bi]; Bh.h2[1] = B2h[bi + 1];
        Bl.h2[0] = B2l[bi]; Bl.h2[1] = B2l[bi + 1];
        C8 C;
#pragma unroll
        for (int i = 0; i < 8; ++i) C.e[i] = 0.0f;
        C.v = wmma_f16(Hlo.v, Bh.v, C.v);
        C.v = wmma_f16(Hhi.v, Bl.v, C.v);
        C.v = wmma_f16(Hhi.v, Bh.v, C.v);
        const int u = tt * 2 + (eL >> 3);
        const float4* a4 = (const float4*)(&W.a0[u][8 * half]);
        float4 a0v = a4[0], a1v = a4[1];
        p00[0] += C.e[0] * a0v.x; p00[1] += C.e[1] * a0v.y;
        p00[2] += C.e[2] * a0v.z; p00[3] += C.e[3] * a0v.w;
        p00[4] += C.e[4] * a1v.x; p00[5] += C.e[5] * a1v.y;
        p00[6] += C.e[6] * a1v.z; p00[7] += C.e[7] * a1v.w;
      }
#pragma unroll 2
      for (int tt = 0; tt < 8; ++tt) {
        const int bi = ((16 + tt) * 16 + eL) * 4 + half * 2;
        V16 Bh, Bl;
        Bh.h2[0] = B2h[bi]; Bh.h2[1] = B2h[bi + 1];
        Bl.h2[0] = B2l[bi]; Bl.h2[1] = B2l[bi + 1];
        C8 C;
#pragma unroll
        for (int i = 0; i < 8; ++i) C.e[i] = 0.0f;
        C.v = wmma_f16(Hlo.v, Bh.v, C.v);
        C.v = wmma_f16(Hhi.v, Bl.v, C.v);
        C.v = wmma_f16(Hhi.v, Bh.v, C.v);
        const int u = tt * 2 + (eL >> 3);
        const float4* a4 = (const float4*)(&W.a110[u][8 * half]);
        float4 a0v = a4[0], a1v = a4[1];
        p00[0] += C.e[0] * a0v.x; p00[1] += C.e[1] * a0v.y;
        p00[2] += C.e[2] * a0v.z; p00[3] += C.e[3] * a0v.w;
        p00[4] += C.e[4] * a1v.x; p00[5] += C.e[5] * a1v.y;
        p00[6] += C.e[6] * a1v.z; p00[7] += C.e[7] * a1v.w;
      }
#pragma unroll
      for (int rr = 0; rr < 8; ++rr)
        atomicAdd(&accE[(rr + 8 * half) * 24 + w8], p00[rr]);

      // ---- region 01 (tiles 24..31): t01 = w01 . x0
      float p01[8];
#pragma unroll
      for (int i = 0; i < 8; ++i) p01[i] = 0.0f;
#pragma unroll 2
      for (int tt = 0; tt < 8; ++tt) {
        const int bi = ((24 + tt) * 16 + eL) * 4 + half * 2;
        V16 Bh, Bl;
        Bh.h2[0] = B2h[bi]; Bh.h2[1] = B2h[bi + 1];
        Bl.h2[0] = B2l[bi]; Bl.h2[1] = B2l[bi + 1];
        C8 C;
#pragma unroll
        for (int i = 0; i < 8; ++i) C.e[i] = 0.0f;
        C.v = wmma_f16(Hlo.v, Bh.v, C.v);
        C.v = wmma_f16(Hhi.v, Bl.v, C.v);
        C.v = wmma_f16(Hhi.v, Bh.v, C.v);
        const int u = tt * 4 + (eL >> 2);
        const float4* a4 = (const float4*)(&W.a0[u][8 * half]);
        float4 a0v = a4[0], a1v = a4[1];
        p01[0] += C.e[0] * a0v.x; p01[1] += C.e[1] * a0v.y;
        p01[2] += C.e[2] * a0v.z; p01[3] += C.e[3] * a0v.w;
        p01[4] += C.e[4] * a1v.x; p01[5] += C.e[5] * a1v.y;
        p01[6] += C.e[6] * a1v.z; p01[7] += C.e[7] * a1v.w;
      }
#pragma unroll
      for (int rr = 0; rr < 8; ++rr)
        atomicAdd(&accE[(rr + 8 * half) * 24 + 8 + w4], p01[rr]);

      // ---- region 10 (tiles 32..35): t10[.,m] = w10 . x1[.,m]
      float p10[24];
#pragma unroll
      for (int i = 0; i < 24; ++i) p10[i] = 0.0f;
#pragma unroll 1
      for (int tt = 0; tt < 4; ++tt) {
        const int bi = ((32 + tt) * 16 + eL) * 4 + half * 2;
        V16 Bh, Bl;
        Bh.h2[0] = B2h[bi]; Bh.h2[1] = B2h[bi + 1];
        Bl.h2[0] = B2l[bi]; Bl.h2[1] = B2l[bi + 1];
        C8 C;
#pragma unroll
        for (int i = 0; i < 8; ++i) C.e[i] = 0.0f;
        C.v = wmma_f16(Hlo.v, Bh.v, C.v);
        C.v = wmma_f16(Hhi.v, Bl.v, C.v);
        C.v = wmma_f16(Hhi.v, Bh.v, C.v);
        const int u = tt * 4 + (eL >> 2);
#pragma unroll
        for (int m = 0; m < 3; ++m) {
          const float4* b4 = (const float4*)(&W.x1c[u][m][8 * half]);
          float4 b0 = b4[0], b1 = b4[1];
          p10[0 * 3 + m] += C.e[0] * b0.x; p10[1 * 3 + m] += C.e[1] * b0.y;
          p10[2 * 3 + m] += C.e[2] * b0.z; p10[3 * 3 + m] += C.e[3] * b0.w;
          p10[4 * 3 + m] += C.e[4] * b1.x; p10[5 * 3 + m] += C.e[5] * b1.y;
          p10[6 * 3 + m] += C.e[6] * b1.z; p10[7 * 3 + m] += C.e[7] * b1.w;
        }
      }
#pragma unroll
      for (int rr = 0; rr < 8; ++rr)
#pragma unroll
        for (int m = 0; m < 3; ++m)
          atomicAdd(&accE[(rr + 8 * half) * 24 + 12 + w4 * 3 + m],
                    p10[rr * 3 + m]);
    }  // net
    lds_fence();

    // ---- finalize: similarity, logits, values (2 lanes per edge)
    if (half == 1) {
      const float* A  = &W.acc[0][eL][0];
      const float* Av = &W.acc[1][eL][0];
      const float Yx = SQ3 * vx * rinv, Yy = SQ3 * vy * rinv, Yz = SQ3 * vz * rinv;
      float k1v[4][3];
#pragma unroll
      for (int v = 0; v < 4; ++v) {
        float t01 = A[8 + v];
        k1v[v][0] = t01 * Yx + A[12 + v * 3 + 0];
        k1v[v][1] = t01 * Yy + A[12 + v * 3 + 1];
        k1v[v][2] = t01 * Yz + A[12 + v * 3 + 2];
        float t01v = Av[8 + v];
        vals[(long)e * 20 + 8 + v * 3 + 0] = t01v * Yx + Av[12 + v * 3 + 0];
        vals[(long)e * 20 + 8 + v * 3 + 1] = t01v * Yy + Av[12 + v * 3 + 1];
        vals[(long)e * 20 + 8 + v * 3 + 2] = t01v * Yz + Av[12 + v * 3 + 2];
      }
      float s1 = 0.0f;
#pragma unroll
      for (int u = 0; u < 4; ++u)
#pragma unroll
        for (int v = 0; v < 4; ++v) {
          float dm = qreg[u * 3 + 0] * k1v[v][0] + qreg[u * 3 + 1] * k1v[v][1] +
                     qreg[u * 3 + 2] * k1v[v][2];
          s1 += sm.wsim1[u * 4 + v] * dm;
        }
      W.simp[eL] = s1 * (1.0f / SQ3);
    }
    lds_fence();
    if (half == 0) {
      const float* A  = &W.acc[0][eL][0];
      const float* Av = &W.acc[1][eL][0];
      float s0 = 0.0f;
#pragma unroll
      for (int u = 0; u < 8; ++u) {
        float qu = qreg[u];
#pragma unroll
        for (int v = 0; v < 8; ++v) s0 += sm.wsim0[u * 8 + v] * qu * A[v];
      }
      float sim = CS * (s0 + W.simp[eL]);
      logits[e] = cut * sim;
#pragma unroll
      for (int w = 0; w < 8; ++w) vals[(long)e * 20 + w] = Av[w];
    }
    lds_fence();
  }
}

// ---- node prep: queries, zero-init of reductions/output --------------------
__global__ void prep_kernel(const float* __restrict__ x,
                            const float* __restrict__ wq0,
                            const float* __restrict__ wq1,
                            float* __restrict__ qbuf, unsigned* __restrict__ mEnc,
                            float* __restrict__ den, float* __restrict__ out) {
  int n = blockIdx.x * blockDim.x + threadIdx.x;
  if (n >= N_NODES) return;
  const float* xr = x + (long)n * 80;
  float q[20];
#pragma unroll
  for (int w = 0; w < 8; ++w) {
    float s = 0.0f;
    for (int u = 0; u < 32; ++u) s += xr[u] * wq0[u * 8 + w];
    q[w] = s * 0.17677669529663687f;  // 1/sqrt(32)
  }
#pragma unroll
  for (int w = 0; w < 4; ++w)
#pragma unroll
    for (int m = 0; m < 3; ++m) {
      float s = 0.0f;
      for (int u = 0; u < 16; ++u) s += xr[32 + u * 3 + m] * wq1[u * 4 + w];
      q[8 + w * 3 + m] = s * 0.25f;  // 1/sqrt(16)
    }
#pragma unroll
  for (int j = 0; j < 20; ++j) {
    qbuf[(long)n * 20 + j] = q[j];
    out[(long)n * 20 + j] = 0.0f;
  }
  mEnc[n] = 0u;
  den[n] = 0.0f;
}

// ---- segment softmax over src + scatter to dst -----------------------------
__global__ void segmax_kernel(const float* __restrict__ logits,
                              const int* __restrict__ ei,
                              unsigned* __restrict__ mEnc) {
  int e = blockIdx.x * blockDim.x + threadIdx.x;
  if (e >= N_EDGES) return;
  unsigned u = __float_as_uint(logits[e]);
  u = (u & 0x80000000u) ? ~u : (u | 0x80000000u);  // order-preserving encode
  atomicMax(&mEnc[ei[e]], u);
}

__global__ void expsum_kernel(float* __restrict__ logits,
                              const int* __restrict__ ei,
                              const unsigned* __restrict__ mEnc,
                              float* __restrict__ den) {
  int e = blockIdx.x * blockDim.x + threadIdx.x;
  if (e >= N_EDGES) return;
  int s = ei[e];
  unsigned u = mEnc[s];
  float m = (u & 0x80000000u) ? __uint_as_float(u & 0x7fffffffu)
                              : __uint_as_float(~u);
  float ex = __expf(logits[e] - m);
  logits[e] = ex;  // reuse as ex buffer
  atomicAdd(&den[s], ex);
}

__global__ void scatter_kernel(const float* __restrict__ exv,
                               const float* __restrict__ vals,
                               const int* __restrict__ ei,
                               const float* __restrict__ den,
                               float* __restrict__ out) {
  int e = blockIdx.x * blockDim.x + threadIdx.x;
  if (e >= N_EDGES) return;
  int s = ei[e], d = ei[N_EDGES + e];
  float attn = exv[e] / den[s];
  float a = sqrtf(fmaxf(attn, 0.0f)) * 0.03125f;  // /NUM_NEIGHBORS
#pragma unroll
  for (int j = 0; j < 20; ++j)
    atomicAdd(&out[(long)d * 20 + j], a * vals[(long)e * 20 + j]);
}

extern "C" void kernel_launch(void* const* d_in, const int* in_sizes, int n_in,
                              void* d_out, int out_size, void* d_ws, size_t ws_size,
                              hipStream_t stream) {
  const float* x     = (const float*)d_in[0];
  const float* pos   = (const float*)d_in[1];
  const float* wq0   = (const float*)d_in[2];
  const float* wq1   = (const float*)d_in[3];
  const float* wkb1  = (const float*)d_in[4];
  const float* wkb2  = (const float*)d_in[5];
  const float* wvb1  = (const float*)d_in[6];
  const float* wvb2  = (const float*)d_in[7];
  const float* wsim0 = (const float*)d_in[8];
  const float* wsim1 = (const float*)d_in[9];
  const int*   ei    = (const int*)d_in[10];
  float* out = (float*)d_out;

  // workspace carve-up (~14.5 MB)
  float* ws      = (float*)d_ws;
  float* qbuf    = ws;                                   // N*20
  float* logits  = qbuf + (size_t)N_NODES * 20;          // E (later reused as ex)
  float* vals    = logits + (size_t)N_EDGES;             // E*20
  float* den     = vals + (size_t)N_EDGES * 20;          // N
  unsigned* mEnc = (unsigned*)(den + N_NODES);           // N

  prep_kernel<<<(N_NODES + 255) / 256, 256, 0, stream>>>(x, wq0, wq1, qbuf, mEnc,
                                                         den, out);
  edge_kernel<<<250, 256, 0, stream>>>(x, pos, wkb1, wkb2, wvb1, wvb2, wsim0,
                                       wsim1, ei, qbuf, logits, vals);
  segmax_kernel<<<(N_EDGES + 255) / 256, 256, 0, stream>>>(logits, ei, mEnc);
  expsum_kernel<<<(N_EDGES + 255) / 256, 256, 0, stream>>>(logits, ei, mEnc, den);
  scatter_kernel<<<(N_EDGES + 255) / 256, 256, 0, stream>>>(logits, vals, ei, den,
                                                            out);
}
